// SGAE_43611097923982
// MI455X (gfx1250) — matrix operand order repaired and checked by reading, not compiled, not run on hardware
//
#include <hip/hip_runtime.h>
#include <hip/hip_bf16.h>

// ---------------------------------------------------------------------------
// Types for CDNA5 WMMA (wave32, v_wmma_f32_16x16x32_bf16)
// ---------------------------------------------------------------------------
typedef __bf16 bf16_t;
typedef __attribute__((ext_vector_type(16))) __bf16 v16bf;
typedef __attribute__((ext_vector_type(8)))  __bf16 v8bf;
typedef __attribute__((ext_vector_type(4)))  __bf16 v4bf;
typedef __attribute__((ext_vector_type(8)))  float  v8f;
typedef __attribute__((ext_vector_type(4)))  float  v4f;

#define TILE_M 128
#define TILE_N 128
#define TILE_K 32
#define LDA_S 40   // padded LDS row stride (bf16 elems); k*2B offsets stay 8B-aligned
#define LDB_S 40

enum { ACT_NONE = 0, ACT_LEAKY = 1, ACT_TANH = 2, ACT_SIG = 3 };

template<int ACT>
__device__ __forceinline__ float apply_act(float v) {
  if (ACT == ACT_LEAKY) return v > 0.f ? v : 0.2f * v;
  if (ACT == ACT_TANH)  return tanhf(v);
  if (ACT == ACT_SIG)   return 1.f / (1.f + __expf(-v));
  return v;
}

// C(MxN) = scaleA * act( A(MxK) @ B + bias ) + scaleB * addend
//   TRANSB=0: B is KxN row-major.  TRANSB=1: B is NxK row-major (C = A @ B^T).
//   scaleA read from scaleA_ptr (device scalar, e.g. alpha) if non-null.
template<int ACT, bool TRANSB>
__global__ __launch_bounds__(256) void gemm_wmma_kernel(
    const float* __restrict__ A, const float* __restrict__ B,
    const float* __restrict__ bias, const float* __restrict__ addend,
    const float* __restrict__ scaleA_ptr, float scaleA_c, float scaleB,
    float* __restrict__ C, int M, int K, int N) {
  __shared__ bf16_t sA[2][TILE_M * LDA_S];
  __shared__ bf16_t sB[2][TILE_N * LDB_S];   // stored as [n][k]

  const int tid  = threadIdx.x;
  const int lane = tid & 31;
  const int wave = tid >> 5;              // 8 waves
  const int wm   = wave & 1;              // 2 waves over M
  const int wn   = wave >> 1;             // 4 waves over N
  const int tileM = blockIdx.y * TILE_M;
  const int tileN = blockIdx.x * TILE_N;
  const bool fullM = (tileM + TILE_M) <= M;
  const bool fullN = (tileN + TILE_N) <= N;

  // ---- staging: global f32 -> LDS bf16, double buffered --------------------
  auto stageA = [&](int k0, bf16_t* dst) {
    if (fullM && (k0 + TILE_K) <= K) {           // fast: vector loads, b64 LDS stores
      v4f r[4];
#pragma unroll
      for (int t = 0; t < 4; ++t) {
        int idx = (t * 256 + tid) * 4;
        int i = idx >> 5, k = idx & 31;
        r[t] = *(const v4f*)&A[(long)(tileM + i) * K + (k0 + k)];
      }
#pragma unroll
      for (int t = 0; t < 4; ++t) {
        int idx = (t * 256 + tid) * 4;
        int i = idx >> 5, k = idx & 31;
        *(v4bf*)&dst[i * LDA_S + k] = __builtin_convertvector(r[t], v4bf);
      }
    } else {                                     // edge: guarded scalar
#pragma unroll
      for (int t = 0; t < 16; ++t) {
        int idx = t * 256 + tid;
        int i = idx >> 5, k = idx & 31;
        int gm = tileM + i, gk = k0 + k;
        float v = (gm < M && gk < K) ? A[(long)gm * K + gk] : 0.f;
        dst[i * LDA_S + k] = (bf16_t)v;
      }
    }
  };
  auto stageB = [&](int k0, bf16_t* dst) {
    if (TRANSB) {
      if (fullN && (k0 + TILE_K) <= K) {
        v4f r[4];
#pragma unroll
        for (int t = 0; t < 4; ++t) {
          int idx = (t * 256 + tid) * 4;
          int n = idx >> 5, k = idx & 31;
          r[t] = *(const v4f*)&B[(long)(tileN + n) * K + (k0 + k)];
        }
#pragma unroll
        for (int t = 0; t < 4; ++t) {
          int idx = (t * 256 + tid) * 4;
          int n = idx >> 5, k = idx & 31;
          *(v4bf*)&dst[n * LDB_S + k] = __builtin_convertvector(r[t], v4bf);
        }
      } else {
#pragma unroll
        for (int t = 0; t < 16; ++t) {
          int idx = t * 256 + tid;
          int n = idx >> 5, k = idx & 31;
          int gn = tileN + n, gk = k0 + k;
          float v = (gn < N && gk < K) ? B[(long)gn * K + gk] : 0.f;
          dst[n * LDB_S + k] = (bf16_t)v;
        }
      }
    } else {
      if (fullN && (k0 + TILE_K) <= K) {         // vector load along n, scatter to [n][k]
        v4f r[4];
#pragma unroll
        for (int t = 0; t < 4; ++t) {
          int idx = (t * 256 + tid) * 4;
          int k = idx >> 7, n = idx & 127;
          r[t] = *(const v4f*)&B[(long)(k0 + k) * N + (tileN + n)];
        }
#pragma unroll
        for (int t = 0; t < 4; ++t) {
          int idx = (t * 256 + tid) * 4;
          int k = idx >> 7, n = idx & 127;
#pragma unroll
          for (int j = 0; j < 4; ++j) dst[(n + j) * LDB_S + k] = (bf16_t)r[t][j];
        }
      } else {
#pragma unroll
        for (int t = 0; t < 16; ++t) {
          int idx = t * 256 + tid;
          int k = idx >> 7, n = idx & 127;
          int gn = tileN + n, gk = k0 + k;
          float v = (gn < N && gk < K) ? B[(long)gk * N + gn] : 0.f;
          dst[n * LDB_S + k] = (bf16_t)v;
        }
      }
    }
  };

  v8f acc[4][2];
#pragma unroll
  for (int t = 0; t < 4; ++t)
#pragma unroll
    for (int u = 0; u < 2; ++u)
#pragma unroll
      for (int r = 0; r < 8; ++r) acc[t][u][r] = 0.f;

  const int lh  = lane >> 4;              // half-wave select
  const int ll  = lane & 15;
  const int aK0 = lh * 8;                 // A frag: K chunks [aK0,+8) and [aK0+16,+8)
  const int bK0 = lh * 8;                 // B frag: K chunks [bK0,+8) and [bK0+16,+8)

  const int ktiles = (K + TILE_K - 1) / TILE_K;
  stageA(0, sA[0]);
  stageB(0, sB[0]);
  __syncthreads();

  for (int kt = 0; kt < ktiles; ++kt) {
    const int cur = kt & 1;
    if (kt + 1 < ktiles) {                // prefetch next tile into other buffer
      stageA((kt + 1) * TILE_K, sA[cur ^ 1]);
      stageB((kt + 1) * TILE_K, sB[cur ^ 1]);
    }
    // ---- fragments per ISA 16-bit layouts --------------------------------
    v16bf afrag[4], bfrag[2];
#pragma unroll
    for (int t = 0; t < 4; ++t) {
      int r = wm * 64 + t * 16 + ll;
      v8bf lo = *(const v8bf*)&sA[cur][r * LDA_S + aK0];
      v8bf hi = *(const v8bf*)&sA[cur][r * LDA_S + aK0 + 16];
      afrag[t] = __builtin_shufflevector(lo, hi, 0,1,2,3,4,5,6,7,8,9,10,11,12,13,14,15);
    }
#pragma unroll
    for (int u = 0; u < 2; ++u) {
      int n = wn * 32 + u * 16 + ll;
      v8bf lo = *(const v8bf*)&sB[cur][n * LDB_S + bK0];
      v8bf hi = *(const v8bf*)&sB[cur][n * LDB_S + bK0 + 16];
      bfrag[u] = __builtin_shufflevector(lo, hi, 0,1,2,3,4,5,6,7,8,9,10,11,12,13,14,15);
    }
#pragma unroll
    for (int t = 0; t < 4; ++t)
#pragma unroll
      for (int u = 0; u < 2; ++u)
        acc[t][u] = __builtin_amdgcn_wmma_f32_16x16x32_bf16(
            false, afrag[t], false, bfrag[u], (short)0, acc[t][u], false, false);
    __syncthreads();                      // staging(next) done + reads(cur) done
  }

  // ---- epilogue: lane = col (ll), VGPR r = row r + 8*lh --------------------
  const float sa = scaleA_ptr ? *scaleA_ptr : scaleA_c;
  const int rbase = lh * 8;
  if (fullM && fullN) {
#pragma unroll
    for (int u = 0; u < 2; ++u) {
      int gn = tileN + wn * 32 + u * 16 + ll;
      float bvv = bias ? bias[gn] : 0.f;
#pragma unroll
      for (int t = 0; t < 4; ++t) {
#pragma unroll
        for (int r = 0; r < 8; ++r) {
          int gm = tileM + wm * 64 + t * 16 + rbase + r;
          long idx = (long)gm * N + gn;
          float o = sa * apply_act<ACT>(acc[t][u][r] + bvv);
          if (addend) o += scaleB * addend[idx];
          C[idx] = o;
        }
      }
    }
  } else {
#pragma unroll
    for (int u = 0; u < 2; ++u) {
      int gn = tileN + wn * 32 + u * 16 + ll;
      if (gn >= N) continue;
      float bvv = bias ? bias[gn] : 0.f;
#pragma unroll
      for (int t = 0; t < 4; ++t) {
#pragma unroll
        for (int r = 0; r < 8; ++r) {
          int gm = tileM + wm * 64 + t * 16 + rbase + r;
          if (gm >= M) continue;
          long idx = (long)gm * N + gn;
          float o = sa * apply_act<ACT>(acc[t][u][r] + bvv);
          if (addend) o += scaleB * addend[idx];
          C[idx] = o;
        }
      }
    }
  }
}

// ---------------------------------------------------------------------------
// Row softmax (in place), one block per row
// ---------------------------------------------------------------------------
__global__ __launch_bounds__(256) void softmax_rows_kernel(float* __restrict__ S, int N) {
  __shared__ float red[256];
  float* p = S + (long)blockIdx.x * N;
  int t = threadIdx.x;
  float m = -3.4e38f;
  for (int j = t; j < N; j += 256) m = fmaxf(m, p[j]);
  red[t] = m; __syncthreads();
  for (int s = 128; s > 0; s >>= 1) { if (t < s) red[t] = fmaxf(red[t], red[t + s]); __syncthreads(); }
  m = red[0]; __syncthreads();
  float sum = 0.f;
  for (int j = t; j < N; j += 256) { float e = __expf(p[j] - m); p[j] = e; sum += e; }
  red[t] = sum; __syncthreads();
  for (int s = 128; s > 0; s >>= 1) { if (t < s) red[t] += red[t + s]; __syncthreads(); }
  float inv = 1.f / red[0];
  for (int j = t; j < N; j += 256) p[j] *= inv;
}

// readout: Z(n,d) -> out(k*d): mean over contiguous segments of n/k rows
__global__ void readout_kernel(const float* __restrict__ Z, float* __restrict__ out,
                               int n, int d, int k) {
  int idx = blockIdx.x * blockDim.x + threadIdx.x;
  if (idx >= k * d) return;
  int ki = idx / d, j = idx % d, seg = n / k;
  float s = 0.f;
  for (int i = 0; i < seg; ++i) s += Z[(long)(ki * seg + i) * d + j];
  out[idx] = s / (float)seg;
}

// Z_l = a*(Xt1+Xt2)/2 + b*(Z1+Z2)/2
__global__ void combine_zl_kernel(const float* __restrict__ a, const float* __restrict__ b,
                                  const float* __restrict__ x1, const float* __restrict__ x2,
                                  const float* __restrict__ z1, const float* __restrict__ z2,
                                  float* __restrict__ out, int n) {
  int i = blockIdx.x * blockDim.x + threadIdx.x;
  if (i < n) out[i] = a[i] * 0.5f * (x1[i] + x2[i]) + b[i] * 0.5f * (z1[i] + z2[i]);
}

// Q[i][c] = (1/(1+||z_i-c||^2)) normalized over c ; z = (Za+Zb)/2
__global__ void qdist_kernel(const float* __restrict__ Za, const float* __restrict__ Zb,
                             const float* __restrict__ centers, float* __restrict__ Q,
                             int n, int d, int k) {
  int i = blockIdx.x * blockDim.x + threadIdx.x;
  if (i >= n) return;
  float z[20];
  for (int j = 0; j < d; ++j) z[j] = 0.5f * (Za[(long)i * d + j] + Zb[(long)i * d + j]);
  float qs[10]; float tot = 0.f;
  for (int c = 0; c < k; ++c) {
    float s = 0.f;
    for (int j = 0; j < d; ++j) { float df = z[j] - centers[c * d + j]; s += df * df; }
    float q = 1.f / (1.f + s);
    qs[c] = q; tot += q;
  }
  float inv = 1.f / tot;
  for (int c = 0; c < k; ++c) Q[(long)i * k + c] = qs[c] * inv;
}

// ---------------------------------------------------------------------------
// Host-side dispatch (only the 5 (ACT, TRANSB) combos actually used)
// ---------------------------------------------------------------------------
static void gemm(hipStream_t s, const float* A, const float* B, const float* bias,
                 const float* addend, const float* saP, float saC, float sb,
                 float* C, int M, int K, int N, int act, bool transB) {
  dim3 grid((N + TILE_N - 1) / TILE_N, (M + TILE_M - 1) / TILE_M), blk(256);
#define GL(AC, TB) gemm_wmma_kernel<AC, TB><<<grid, blk, 0, s>>>(A, B, bias, addend, saP, saC, sb, C, M, K, N)
  if (!transB) {
    switch (act) {
      case ACT_LEAKY: GL(ACT_LEAKY, false); break;
      case ACT_TANH:  GL(ACT_TANH,  false); break;
      default:        GL(ACT_NONE,  false); break;
    }
  } else {
    if (act == ACT_SIG) GL(ACT_SIG, true);
    else                GL(ACT_NONE, true);
  }
#undef GL
}

extern "C" void kernel_launch(void* const* d_in, const int* in_sizes, int n_in,
                              void* d_out, int out_size, void* d_ws, size_t ws_size,
                              hipStream_t stream) {
  (void)in_sizes; (void)n_in; (void)out_size; (void)ws_size;
  const int N = 4000, NIN = 3000, NZ = 20, KC = 10;

  const float* X1 = (const float*)d_in[0];
  const float* Am = (const float*)d_in[1];
  const float* X2 = (const float*)d_in[2];
  const float* Ad = (const float*)d_in[3];
  // ae_enc
  const float* eW1 = (const float*)d_in[4];  const float* eb1 = (const float*)d_in[5];
  const float* eW2 = (const float*)d_in[6];  const float* eb2 = (const float*)d_in[7];
  const float* eW3 = (const float*)d_in[8];  const float* eb3 = (const float*)d_in[9];
  const float* eWz = (const float*)d_in[10]; const float* ebz = (const float*)d_in[11];
  // ae_dec
  const float* dW1 = (const float*)d_in[12]; const float* db1 = (const float*)d_in[13];
  const float* dW2 = (const float*)d_in[14]; const float* db2 = (const float*)d_in[15];
  const float* dW3 = (const float*)d_in[16]; const float* db3 = (const float*)d_in[17];
  const float* dWx = (const float*)d_in[18]; const float* dbx = (const float*)d_in[19];
  // gae enc / dec
  const float* ge1 = (const float*)d_in[20]; const float* ge2 = (const float*)d_in[21];
  const float* ge3 = (const float*)d_in[22];
  const float* gd1 = (const float*)d_in[23]; const float* gd2 = (const float*)d_in[24];
  const float* gd3 = (const float*)d_in[25];
  const float* av  = (const float*)d_in[26]; const float* bv  = (const float*)d_in[27];
  const float* alpha = (const float*)d_in[28];
  const float* centers = (const float*)d_in[29];

  // ---- output layout (flat, reference return order) ----------------------
  float* out = (float*)d_out;
  float* Xhat  = out;                       // 4000x3000
  float* Zhat  = out + 12000000;            // 4000x3000
  float* Ahat  = out + 24000000;            // 4000x4000
  float* simO  = out + 40000000;            // 4000x4000
  float* Xt1   = out + 56000000;            // 4000x20
  float* Xt2   = out + 56080000;
  float* ZtAe1 = out + 56160000;            // 200
  float* ZtAe2 = out + 56160200;
  float* Zig1  = out + 56160400;            // 4000x20
  float* Zig2  = out + 56240400;
  float* ZtIg1 = out + 56320400;            // 200
  float* ZtIg2 = out + 56320600;
  float* Q0    = out + 56320800;            // 4000x10 each
  float* Q1    = out + 56360800;
  float* Q2    = out + 56400800;
  float* Zo    = out + 56440800;            // 4000x20

  // ---- workspace ----------------------------------------------------------
  float* ws   = (float*)d_ws;
  float* buf0 = ws;                         // 4000x512
  float* buf1 = ws + 2048000;               // 4000x512
  float* zl   = ws + 4096000;               // 4000x20
  float* zl2  = ws + 4176000;               // 4000x20
  float* S    = ws + 4256000;               // 4000x4000 (reused as 4000x3000 temp)

  const float* NP = nullptr;
  hipStream_t s = stream;

  // ================= GAE encoder view 1 (X1, Am) ==========================
  gemm(s, X1, ge1, NP, NP, NP, 1.f, 0.f, buf0, N, NIN, 128, ACT_NONE, false);
  gemm(s, Am, buf0, NP, NP, NP, 1.f, 0.f, buf1, N, N, 128, ACT_TANH, false);
  gemm(s, buf1, ge2, NP, NP, NP, 1.f, 0.f, buf0, N, 128, 256, ACT_NONE, false);
  gemm(s, Am, buf0, NP, NP, NP, 1.f, 0.f, buf1, N, N, 256, ACT_TANH, false);
  gemm(s, buf1, ge3, NP, NP, NP, 1.f, 0.f, buf0, N, 256, NZ, ACT_NONE, false);
  gemm(s, Am, buf0, NP, NP, NP, 1.f, 0.f, Zig1, N, N, NZ, ACT_NONE, false);
  // sim pass 1: 0.5 * sigmoid(Z1 @ Z1^T)
  gemm(s, Zig1, Zig1, NP, NP, NP, 0.5f, 0.f, simO, N, NZ, N, ACT_SIG, true);

  // ================= GAE encoder view 2 (X2, Ad) ==========================
  gemm(s, X2, ge1, NP, NP, NP, 1.f, 0.f, buf0, N, NIN, 128, ACT_NONE, false);
  gemm(s, Ad, buf0, NP, NP, NP, 1.f, 0.f, buf1, N, N, 128, ACT_TANH, false);
  gemm(s, buf1, ge2, NP, NP, NP, 1.f, 0.f, buf0, N, 128, 256, ACT_NONE, false);
  gemm(s, Ad, buf0, NP, NP, NP, 1.f, 0.f, buf1, N, N, 256, ACT_TANH, false);
  gemm(s, buf1, ge3, NP, NP, NP, 1.f, 0.f, buf0, N, 256, NZ, ACT_NONE, false);
  gemm(s, Ad, buf0, NP, NP, NP, 1.f, 0.f, Zig2, N, N, NZ, ACT_NONE, false);
  // sim pass 2: sim = 0.5*sigmoid(Z2 @ Z2^T) + sim
  gemm(s, Zig2, Zig2, NP, simO, NP, 0.5f, 1.f, simO, N, NZ, N, ACT_SIG, true);

  // ================= AE encoders ==========================================
  gemm(s, X1, eW1, eb1, NP, NP, 1.f, 0.f, buf0, N, NIN, 128, ACT_LEAKY, false);
  gemm(s, buf0, eW2, eb2, NP, NP, 1.f, 0.f, buf1, N, 128, 256, ACT_LEAKY, false);
  gemm(s, buf1, eW3, eb3, NP, NP, 1.f, 0.f, buf0, N, 256, 512, ACT_LEAKY, false);
  gemm(s, buf0, eWz, ebz, NP, NP, 1.f, 0.f, Xt1, N, 512, NZ, ACT_NONE, false);

  gemm(s, X2, eW1, eb1, NP, NP, 1.f, 0.f, buf0, N, NIN, 128, ACT_LEAKY, false);
  gemm(s, buf0, eW2, eb2, NP, NP, 1.f, 0.f, buf1, N, 128, 256, ACT_LEAKY, false);
  gemm(s, buf1, eW3, eb3, NP, NP, 1.f, 0.f, buf0, N, 256, 512, ACT_LEAKY, false);
  gemm(s, buf0, eWz, ebz, NP, NP, 1.f, 0.f, Xt2, N, 512, NZ, ACT_NONE, false);

  // ================= readouts =============================================
  readout_kernel<<<1, 256, 0, s>>>(Zig1, ZtIg1, N, NZ, KC);
  readout_kernel<<<1, 256, 0, s>>>(Zig2, ZtIg2, N, NZ, KC);
  readout_kernel<<<1, 256, 0, s>>>(Xt1, ZtAe1, N, NZ, KC);
  readout_kernel<<<1, 256, 0, s>>>(Xt2, ZtAe2, N, NZ, KC);

  // ================= fusion + attention ===================================
  combine_zl_kernel<<<(N * NZ + 255) / 256, 256, 0, s>>>(av, bv, Xt1, Xt2, Zig1, Zig2, zl, N * NZ);
  gemm(s, Am, zl, NP, NP, NP, 1.f, 0.f, zl2, N, N, NZ, ACT_NONE, false);
  gemm(s, zl2, zl2, NP, NP, NP, 1.f, 0.f, S, N, NZ, N, ACT_NONE, true);   // scores
  softmax_rows_kernel<<<N, 256, 0, s>>>(S, N);
  // Z = alpha * (softmax(S) @ zl2) + zl2
  gemm(s, S, zl2, NP, zl2, alpha, 1.f, 1.f, Zo, N, N, NZ, ACT_NONE, false);

  // ================= AE decoder ===========================================
  gemm(s, Zo, dW1, db1, NP, NP, 1.f, 0.f, buf0, N, NZ, 512, ACT_LEAKY, false);
  gemm(s, buf0, dW2, db2, NP, NP, 1.f, 0.f, buf1, N, 512, 256, ACT_LEAKY, false);
  gemm(s, buf1, dW3, db3, NP, NP, 1.f, 0.f, buf0, N, 256, 128, ACT_LEAKY, false);
  gemm(s, buf0, dWx, dbx, NP, NP, 1.f, 0.f, Xhat, N, 128, NIN, ACT_NONE, false);

  // ================= GAE decoder ==========================================
  gemm(s, Zo, gd1, NP, NP, NP, 1.f, 0.f, buf0, N, NZ, 256, ACT_NONE, false);
  gemm(s, Am, buf0, NP, NP, NP, 1.f, 0.f, buf1, N, N, 256, ACT_TANH, false);
  gemm(s, buf1, gd2, NP, NP, NP, 1.f, 0.f, buf0, N, 256, 128, ACT_NONE, false);
  gemm(s, Am, buf0, NP, NP, NP, 1.f, 0.f, buf1, N, N, 128, ACT_TANH, false);
  gemm(s, buf1, gd3, NP, NP, NP, 1.f, 0.f, S, N, 128, NIN, ACT_NONE, false);   // reuse S
  gemm(s, Am, S, NP, NP, NP, 1.f, 0.f, Zhat, N, N, NIN, ACT_TANH, false);      // 96 GFLOP
  // A_hat = sim + sigmoid(Zhat @ Zhat^T)                                      // 96 GFLOP
  gemm(s, Zhat, Zhat, NP, simO, NP, 1.f, 1.f, Ahat, N, NIN, N, ACT_SIG, true);

  // ================= Q distributions ======================================
  qdist_kernel<<<(N + 255) / 256, 256, 0, s>>>(Zo, Zo, centers, Q0, N, NZ, KC);
  qdist_kernel<<<(N + 255) / 256, 256, 0, s>>>(Xt1, Xt2, centers, Q1, N, NZ, KC);
  qdist_kernel<<<(N + 255) / 256, 256, 0, s>>>(Zig1, Zig2, centers, Q2, N, NZ, KC);
}
